// GraphSAGE_79474074845319
// MI455X (gfx1250) — matrix operand order repaired and checked by reading, not compiled, not run on hardware
//
#include <hip/hip_runtime.h>
#include <hip/hip_bf16.h>

typedef _Float16 h16;
typedef __attribute__((ext_vector_type(16))) _Float16 v16h;
typedef __attribute__((ext_vector_type(8)))  _Float16 v8h;
typedef __attribute__((ext_vector_type(8)))  float    v8f;

#define NB 8
#define NN 1024
#define NJ 64
#define ND 64
#define NH 128
#define NEG_SLOPE 0.2f
#define S1_WAVES 8          // j-split waves per stage-1 block

// Wave-local LDS fence: lH1 staging slices are wave-private, so a workgroup
// barrier is unnecessary (and would deadlock under divergent j counts).
// DS ops are in-order per wave; s_wait_dscnt 0 guarantees the transpose
// stores have landed before re-reads.
__device__ __forceinline__ void wave_lds_fence() {
    asm volatile("s_wait_dscnt 0x0" ::: "memory");
}

// ---------------------------------------------------------------------------
// WMMA A-layout (16x32 f16, wave32): lane L holds row m=L%16.
// element i -> K = kc*32 + (L>=16 ? 8:0) + (i<8 ? i : 16+(i-8))
// ---------------------------------------------------------------------------
__device__ __forceinline__ v16h loadA_glb(const float* __restrict__ row, int halfsel, int kc) {
    const int off = kc * 32 + halfsel * 8;
    v16h a;
#pragma unroll
    for (int i = 0; i < 8; ++i) {
        a[i]     = (h16)row[off + i];
        a[i + 8] = (h16)row[off + 16 + i];
    }
    return a;
}

__device__ __forceinline__ v16h loadA_lds(const h16* base, int ld, int m, int halfsel, int kc) {
    const h16* p = base + m * ld + kc * 32 + halfsel * 8;
    v8h lo = *(const v8h*)(p);
    v8h hi = *(const v8h*)(p + 16);
    v16h a;
#pragma unroll
    for (int i = 0; i < 8; ++i) { a[i] = lo[i]; a[i + 8] = hi[i]; }
    return a;
}

// ---------------------------------------------------------------------------
// Stage weights into LDS in WMMA B layout:
//  W1 (HxD=128x64): tile (kc,ht): lane L: column n=L%16, halfword s: row k=(s&7)+8*(L/16)+16*(s>>3)
//  -> W1[ht*16+n][kc*32+k].  Same scheme for W2 (DxH).
// ---------------------------------------------------------------------------
__device__ __forceinline__ void load_weights(const float* __restrict__ w1, const float* __restrict__ b1,
                                             const float* __restrict__ w2, const float* __restrict__ b2,
                                             h16* lW1, h16* lW2, float* lb1, float* lb2,
                                             int tid, int nthreads) {
    for (int idx = tid; idx < 8192; idx += nthreads) {      // W1: (kc*8+ht)*512 + lane*16 + s
        int n = (idx >> 4) & 15, k = (idx & 7) + ((idx >> 8) & 1) * 8 + ((idx >> 3) & 1) * 16, t = (idx >> 9) & 7, c = idx >> 12;
        lW1[idx] = (h16)w1[(t * 16 + n) * ND + (c * 32 + k)];
    }
    for (int idx = tid; idx < 8192; idx += nthreads) {      // W2: (kc*4+dt)*512 + lane*16 + s
        int n = (idx >> 4) & 15, k = (idx & 7) + ((idx >> 8) & 1) * 8 + ((idx >> 3) & 1) * 16, dt = (idx >> 9) & 3, kc = idx >> 11;
        lW2[idx] = (h16)w2[(dt * 16 + n) * NH + (kc * 32 + k)];
    }
    for (int i = tid; i < NH; i += nthreads) lb1[i] = b1[i];
    for (int i = tid; i < ND; i += nthreads) lb2[i] = b2[i];
}

// ---------------------------------------------------------------------------
// 16-row MLP tile: [16x64] -> GEMM1 -> (+b1, ELU) -> GEMM2 -> out[4] + b2.
// Accumulators start at inline 0 (VOP3P SRC2=0); biases folded into epilogues.
// B-tiles are software-pipelined one step ahead; GEMM2 A-chunks are register
// cached. lH1 must be a wave-private 16x128 f16 staging buffer.
// ---------------------------------------------------------------------------
__device__ __forceinline__ void mlp16(v16h A0, v16h A1,
                                      const h16* lW1, const h16* lW2,
                                      const float* lb1, const float* lb2,
                                      h16* lH1, int lane, v8f out[4]) {
    const int m = lane & 15, hsel = lane >> 4, nc = lane & 15;

    // ---- GEMM1: 16x64 @ 64x128, B double-buffered one ht ahead ----
    v16h B0 = *(const v16h*)(lW1 + lane * 16);              // (kc=0, ht=0)
    v16h B1 = *(const v16h*)(lW1 + 8 * 512 + lane * 16);    // (kc=1, ht=0)
#pragma unroll
    for (int ht = 0; ht < 8; ++ht) {
        const v16h B0c = B0, B1c = B1;
        if (ht < 7) {
            B0 = *(const v16h*)(lW1 + (ht + 1) * 512 + lane * 16);
            B1 = *(const v16h*)(lW1 + (ht + 9) * 512 + lane * 16);
        }
        v8f c1 = {};                                        // -> v_wmma ..., 0
        c1 = __builtin_amdgcn_wmma_f32_16x16x32_f16(false, A0, false, B0c, (short)0, c1, false, false);
        c1 = __builtin_amdgcn_wmma_f32_16x16x32_f16(false, A1, false, B1c, (short)0, c1, false, false);
        const float bias = lb1[ht * 16 + nc];
#pragma unroll
        for (int v = 0; v < 8; ++v) {                       // bias + ELU, C->A transpose via LDS
            float val = c1[v] + bias;
            val = val > 0.0f ? val : (__expf(val) - 1.0f);
            lH1[(v + 8 * hsel) * NH + ht * 16 + nc] = (h16)val;
        }
    }
    wave_lds_fence();

    // ---- GEMM2: 16x128 @ 128x64; A-chunks register-cached, B pipelined ----
    v16h Ah[4];
#pragma unroll
    for (int kc = 0; kc < 4; ++kc) Ah[kc] = loadA_lds(lH1, NH, m, hsel, kc);

    v16h Bw = *(const v16h*)(lW2 + lane * 16);              // (kc=0, dt=0)
    v8f c2 = {};
#pragma unroll
    for (int q = 0; q < 16; ++q) {                          // q = dt*4 + kc
        const int dt = q >> 2, kc = q & 3;
        const v16h Bc = Bw;
        if (q < 15) {
            const int qn = q + 1, dtn = qn >> 2, kcn = qn & 3;
            Bw = *(const v16h*)(lW2 + (kcn * 4 + dtn) * 512 + lane * 16);
        }
        c2 = __builtin_amdgcn_wmma_f32_16x16x32_f16(false, Ah[kc], false, Bc, (short)0, c2, false, false);
        if (kc == 3) {
            const float bias = lb2[dt * 16 + nc];
#pragma unroll
            for (int v = 0; v < 8; ++v) c2[v] += bias;
            out[dt] = c2;
            c2 = (v8f){};
        }
    }
}

// ---------------------------------------------------------------------------
// Stage 1: fused  sum_j M*(MLP_f(x_tm1*rpow))  +  agg_pre*cnt  -> xwork
// One block per (b, 16-row n-tile); 8 waves split the j loop (j = w mod 8),
// partial accumulators combined through LDS by wave 0.
// ---------------------------------------------------------------------------
__global__ __launch_bounds__(32 * S1_WAVES) void k_stage1(
        const float* __restrict__ x,
        const int* __restrict__ sarr, const int* __restrict__ earr,
        const float* __restrict__ rptr,
        const float* __restrict__ f1w, const float* __restrict__ f1b,
        const float* __restrict__ f2w, const float* __restrict__ f2b,
        float* __restrict__ xwork) {
    __shared__ __attribute__((aligned(32))) h16 lW1[8192];
    __shared__ __attribute__((aligned(32))) h16 lW2[8192];
    __shared__ __attribute__((aligned(16))) h16 lH1[S1_WAVES * 16 * NH];
    __shared__ float lb1[NH];
    __shared__ float lb2[ND];
    __shared__ float redAcc[S1_WAVES * 16 * ND];
    __shared__ float redCnt[S1_WAVES * 16];

    const int tid  = threadIdx.x;
    const int lane = tid & 31;
    const int wave = tid >> 5;
    const int b  = blockIdx.x >> 6;
    const int n0 = (blockIdx.x & 63) << 4;
    const float rdisc = *rptr;

    load_weights(f1w, f1b, f2w, f2b, lW1, lW2, lb1, lb2, tid, 32 * S1_WAVES);
    __syncthreads();

    const int m = lane & 15, hsel = lane >> 4, nc = lane & 15;
    const int trow = n0 + m;
    const int src = (trow + NN - 1) & (NN - 1);             // roll(x, 1)
    const float* xr = x + ((size_t)b * NN + src) * ND;
    const v16h A0 = loadA_glb(xr, hsel, 0);
    const v16h A1 = loadA_glb(xr, hsel, 1);
    h16* myH = lH1 + wave * 16 * NH;

    v8f acc[4];
    float cnt[8];
#pragma unroll
    for (int dt = 0; dt < 4; ++dt)
#pragma unroll
        for (int v = 0; v < 8; ++v) acc[dt][v] = 0.0f;
#pragma unroll
    for (int v = 0; v < 8; ++v) cnt[v] = 0.0f;

    for (int j = wave; j < NJ; j += S1_WAVES) {
        const int sj = sarr[b * NJ + j];                    // scalar-uniform per wave
        const int ej = earr[b * NJ + j];
        if (ej < n0 || sj >= n0 + 15) continue;             // tile fully outside (s, e]

        const float rp = (trow > sj) ? __powf(rdisc, (float)(trow - sj)) : 1.0f;
        const h16 hscale = (h16)rp;
        v16h As0 = A0, As1 = A1;
#pragma unroll
        for (int i = 0; i < 16; ++i) { As0[i] *= hscale; As1[i] *= hscale; }

        v8f out[4];
        mlp16(As0, As1, lW1, lW2, lb1, lb2, myH, lane, out);
#pragma unroll
        for (int v = 0; v < 8; ++v) {
            const int tr = n0 + v + 8 * hsel;
            const float mf = ((tr > sj) && (tr <= ej)) ? 1.0f : 0.0f;
            cnt[v] += mf;
#pragma unroll
            for (int dt = 0; dt < 4; ++dt) acc[dt][v] += mf * out[dt][v];
        }
    }

    // combine partial accumulators across waves
    float* myred = redAcc + wave * 16 * ND;
#pragma unroll
    for (int dt = 0; dt < 4; ++dt)
#pragma unroll
        for (int v = 0; v < 8; ++v)
            myred[(v + 8 * hsel) * ND + dt * 16 + nc] = acc[dt][v];
    if (nc == 0) {
#pragma unroll
        for (int v = 0; v < 8; ++v) redCnt[wave * 16 + v + 8 * hsel] = cnt[v];
    }
    __syncthreads();
    if (wave != 0) return;

    for (int w = 1; w < S1_WAVES; ++w) {
        const float* rp_ = redAcc + w * 16 * ND;
#pragma unroll
        for (int dt = 0; dt < 4; ++dt)
#pragma unroll
            for (int v = 0; v < 8; ++v)
                acc[dt][v] += rp_[(v + 8 * hsel) * ND + dt * 16 + nc];
#pragma unroll
        for (int v = 0; v < 8; ++v) cnt[v] += redCnt[w * 16 + v + 8 * hsel];
    }

    // agg_pre on unshifted x
    const float* xr0 = x + ((size_t)b * NN + trow) * ND;
    const v16h P0 = loadA_glb(xr0, hsel, 0);
    const v16h P1 = loadA_glb(xr0, hsel, 1);
    v8f pre[4];
    mlp16(P0, P1, lW1, lW2, lb1, lb2, myH, lane, pre);

#pragma unroll
    for (int dt = 0; dt < 4; ++dt)
#pragma unroll
        for (int v = 0; v < 8; ++v) {
            const int tr = n0 + v + 8 * hsel;
            const size_t o = ((size_t)b * NN + tr) * ND + dt * 16 + nc;
            xwork[o] = (cnt[v] > 0.0f) ? (pre[dt][v] * cnt[v] + acc[dt][v]) : x[o];
        }
}

// ---------------------------------------------------------------------------
// Stage 2: gather feat_at_s, mix, MLP_z, write job rows [0,J) of xwork.
// One single-wave block per batch b.
// ---------------------------------------------------------------------------
__global__ __launch_bounds__(32) void k_stage2(float* __restrict__ xwork,
                                               const int* __restrict__ sarr,
                                               const int* __restrict__ earr,
                                               const float* __restrict__ rptr,
                                               const float* __restrict__ z1w, const float* __restrict__ z1b,
                                               const float* __restrict__ z2w, const float* __restrict__ z2b) {
    __shared__ __attribute__((aligned(32))) h16 lW1[8192];
    __shared__ __attribute__((aligned(32))) h16 lW2[8192];
    __shared__ __attribute__((aligned(16))) h16 lH1[16 * NH];
    __shared__ __attribute__((aligned(16))) h16 lXm[NJ * ND];
    __shared__ float lb1[NH];
    __shared__ float lb2[ND];
    __shared__ float feat[NJ * ND];
    __shared__ float colsum[ND];

    const int lane = threadIdx.x;
    const int b = blockIdx.x;
    const float rdisc = *rptr;

    load_weights(z1w, z1b, z2w, z2b, lW1, lW2, lb1, lb2, lane, 32);

    for (int idx = lane; idx < NJ * ND; idx += 32) {        // gather feat_at_s
        int j = idx >> 6, d = idx & 63;
        int sj = sarr[b * NJ + j];
        sj = sj < 0 ? 0 : (sj > NN - 1 ? NN - 1 : sj);
        feat[idx] = xwork[((size_t)b * NN + sj) * ND + d];
    }
    __syncthreads();

    float Kf = 0.0f;
    for (int j = 0; j < NJ; ++j) Kf += (sarr[b * NJ + j] <= earr[b * NJ + j]) ? 1.0f : 0.0f;
    if (Kf < 1.0f) Kf = 1.0f;

    for (int d = lane; d < ND; d += 32) {
        float s = 0.0f;
        for (int j = 0; j < NJ; ++j)
            if (sarr[b * NJ + j] <= earr[b * NJ + j]) s += feat[j * ND + d];
        colsum[d] = s;
    }
    __syncthreads();

    for (int idx = lane; idx < NJ * ND; idx += 32) {        // mixed
        int d = idx & 63;
        float f = feat[idx];
        lXm[idx] = (h16)((f + (colsum[d] - f) * rdisc) / Kf);
    }
    __syncthreads();

    const int m = lane & 15, hsel = lane >> 4, nc = lane & 15;
#pragma unroll
    for (int t = 0; t < 4; ++t) {
        const int j0 = t * 16;
        v16h A0 = loadA_lds(lXm + j0 * ND, ND, m, hsel, 0);
        v16h A1 = loadA_lds(lXm + j0 * ND, ND, m, hsel, 1);
        v8f out[4];
        mlp16(A0, A1, lW1, lW2, lb1, lb2, lH1, lane, out);
#pragma unroll
        for (int dt = 0; dt < 4; ++dt)
#pragma unroll
            for (int v = 0; v < 8; ++v) {
                const int j = j0 + v + 8 * hsel;
                if (sarr[b * NJ + j] <= earr[b * NJ + j])   // vm ? mlp_out : keep x row j
                    xwork[((size_t)b * NN + j) * ND + dt * 16 + nc] = out[dt][v];
            }
        __syncthreads();
    }
}

// ---------------------------------------------------------------------------
// Stage 3: MLP_p over all rows + leaky_relu -> d_out.
// 4 waves per block, one 16-row tile each, shared LDS weights.
// ---------------------------------------------------------------------------
__global__ __launch_bounds__(128) void k_stage3(const float* __restrict__ xwork,
                                                const float* __restrict__ p1w, const float* __restrict__ p1b,
                                                const float* __restrict__ p2w, const float* __restrict__ p2b,
                                                float* __restrict__ outp) {
    __shared__ __attribute__((aligned(32))) h16 lW1[8192];
    __shared__ __attribute__((aligned(32))) h16 lW2[8192];
    __shared__ __attribute__((aligned(16))) h16 lH1[4 * 16 * NH];
    __shared__ float lb1[NH];
    __shared__ float lb2[ND];

    const int tid  = threadIdx.x;
    const int lane = tid & 31;
    const int wave = tid >> 5;

    load_weights(p1w, p1b, p2w, p2b, lW1, lW2, lb1, lb2, tid, 128);
    __syncthreads();

    const int tile = blockIdx.x * 4 + wave;
    const int b  = tile >> 6;
    const int n0 = (tile & 63) << 4;

    const int m = lane & 15, hsel = lane >> 4, nc = lane & 15;
    const float* xr = xwork + ((size_t)b * NN + n0 + m) * ND;
    const v16h A0 = loadA_glb(xr, hsel, 0);
    const v16h A1 = loadA_glb(xr, hsel, 1);
    v8f out[4];
    mlp16(A0, A1, lW1, lW2, lb1, lb2, lH1 + wave * 16 * NH, lane, out);

#pragma unroll
    for (int dt = 0; dt < 4; ++dt)
#pragma unroll
        for (int v = 0; v < 8; ++v) {
            float val = out[dt][v];
            val = val > 0.0f ? val : NEG_SLOPE * val;
            outp[((size_t)b * NN + n0 + v + 8 * hsel) * ND + dt * 16 + nc] = val;
        }
}

// ---------------------------------------------------------------------------
extern "C" void kernel_launch(void* const* d_in, const int* in_sizes, int n_in,
                              void* d_out, int out_size, void* d_ws, size_t ws_size,
                              hipStream_t stream) {
    const float* x    = (const float*)d_in[0];
    const int*   s    = (const int*)d_in[1];
    const int*   e    = (const int*)d_in[2];
    // d_in[3] opes_appertain, d_in[4] batch_idxes: unused by the forward math
    const float* r    = (const float*)d_in[5];
    const float* f1w  = (const float*)d_in[6];
    const float* f1b  = (const float*)d_in[7];
    const float* f2w  = (const float*)d_in[8];
    const float* f2b  = (const float*)d_in[9];
    const float* z1w  = (const float*)d_in[10];
    const float* z1b  = (const float*)d_in[11];
    const float* z2w  = (const float*)d_in[12];
    const float* z2b  = (const float*)d_in[13];
    const float* p1w  = (const float*)d_in[14];
    const float* p1b  = (const float*)d_in[15];
    const float* p2w  = (const float*)d_in[16];
    const float* p2b  = (const float*)d_in[17];

    float* xwork = (float*)d_ws;                  // [B,N,D] f32 = 2 MB
    float* outp  = (float*)d_out;

    k_stage1<<<NB * (NN / 16), 32 * S1_WAVES, 0, stream>>>(x, s, e, r, f1w, f1b, f2w, f2b, xwork);
    k_stage2<<<NB, 32, 0, stream>>>(xwork, s, e, r, z1w, z1b, z2w, z2b);
    k_stage3<<<NB * (NN / 16) / 4, 128, 0, stream>>>(xwork, p1w, p1b, p2w, p2b, outp);
}
